// HierarchicalFlowAnchor_35390530519881
// MI455X (gfx1250) — matrix operand order repaired and checked by reading, not compiled, not run on hardware
//
#include <hip/hip_runtime.h>
#include <hip/hip_bf16.h>

// ---------------------------------------------------------------------------
// HierarchicalFlowAnchor for MI455X (gfx1250, wave32, WMMA bf16 + TDM + async)
// ---------------------------------------------------------------------------

typedef __attribute__((ext_vector_type(16))) __bf16   v16bf;
typedef __attribute__((ext_vector_type(8)))  float    v8f;
typedef __attribute__((ext_vector_type(4)))  unsigned u32x4;
typedef __attribute__((ext_vector_type(8)))  unsigned u32x8;

#define B_    4
#define S_    1024
#define D_    512
#define H_    16
#define HD_   32
#define NTOK  (B_ * S_)
#define KPAD  520              // 512 + 8 bf16 (16B) row skew -> no LDS bank conflicts
#define ATT_SCALE 0.17677669529663687f   // 1/sqrt(32)

// ---------------------------------------------------------------------------
// gfx1250 helpers: async global->LDS copy (ASYNCcnt) and TDM 2D tile load
// ---------------------------------------------------------------------------
__device__ __forceinline__ unsigned lds_off_u32(const void* p) {
  // generic shared pointer: low 32 bits are the LDS byte offset
  return (unsigned)(unsigned long long)(size_t)p;
}

__device__ __forceinline__ void async_copy_b128(unsigned lds_off, const void* gsrc) {
  asm volatile("global_load_async_to_lds_b128 %0, %1, off"
               :: "v"(lds_off), "v"((unsigned long long)(size_t)gsrc)
               : "memory");
}

__device__ __forceinline__ void async_wait0() {
  asm volatile("s_wait_asynccnt 0x0" ::: "memory");
}

// TDM: load a (rows x 512) bf16 2D tile into LDS with a 16B pad every row
// (pad_interval = 256 dwords = one 1KB row, pad_amount = 4 dwords = 16B)
__device__ __forceinline__ void tdm_load_tile16(unsigned lds_off, const void* gsrc) {
  unsigned long long ga = (unsigned long long)(size_t)gsrc;
  u32x4 g0;
  g0[0] = 1u;                                   // count=1, user descriptor
  g0[1] = lds_off;                              // lds_addr
  g0[2] = (unsigned)ga;                         // global_addr[31:0]
  g0[3] = (unsigned)(ga >> 32) | (2u << 30);    // global_addr[56:32] | type=2
  u32x8 g1;
  g1[0] = 0x07D10000u;            // data_size=2B | pad_enable | interval=256dw | amount=4dw
  g1[1] = (unsigned)(D_ << 16);   // tensor_dim0 = 512            (bits 79:48 lo)
  g1[2] = (unsigned)((NTOK & 0xffff) << 16);    // tensor_dim1 = 4096 (bits 111:80 lo)
  g1[3] = (unsigned)(D_ << 16);   // tile_dim0 = 512              (bits 127:112)
  g1[4] = 16u;                    // tile_dim1 = 16, tile_dim2 = 0
  g1[5] = (unsigned)D_;           // tensor_dim0_stride = 512     (bits 207:160 lo)
  g1[6] = 0u;
  g1[7] = 0u;
  asm volatile("tensor_load_to_lds %0, %1" :: "s"(g0), "s"(g1) : "memory");
  __builtin_amdgcn_s_wait_tensorcnt(0);
}

// ---------------------------------------------------------------------------
// Kernel 1: LayerNorm, one wave (32 lanes) per token, output bf16
// ---------------------------------------------------------------------------
__global__ void ln_kernel(const float* __restrict__ x, const float* __restrict__ g,
                          const float* __restrict__ bb, __bf16* __restrict__ xnb) {
  int wave = (blockIdx.x * blockDim.x + threadIdx.x) >> 5;
  int lane = threadIdx.x & 31;
  if (wave >= NTOK) return;
  const float* row = x + (size_t)wave * D_;
  float vals[16];
  float s = 0.f;
#pragma unroll
  for (int i = 0; i < 16; ++i) { vals[i] = row[lane * 16 + i]; s += vals[i]; }
#pragma unroll
  for (int off = 16; off > 0; off >>= 1) s += __shfl_xor(s, off, 32);
  float mu = s * (1.0f / (float)D_);
  float var = 0.f;
#pragma unroll
  for (int i = 0; i < 16; ++i) { float d = vals[i] - mu; var += d * d; }
#pragma unroll
  for (int off = 16; off > 0; off >>= 1) var += __shfl_xor(var, off, 32);
  float rstd = rsqrtf(var * (1.0f / (float)D_) + 1e-5f);
  __bf16* orow = xnb + (size_t)wave * D_;
#pragma unroll
  for (int i = 0; i < 16; ++i) {
    int c = lane * 16 + i;
    orow[c] = (__bf16)((vals[i] - mu) * rstd * g[c] + bb[c]);
  }
}

// ---------------------------------------------------------------------------
// Kernel 2: weight f32 [K][N] -> bf16 transposed [N][K] (B-fragment friendly)
// ---------------------------------------------------------------------------
__global__ void wconv_kernel(const float* __restrict__ W, __bf16* __restrict__ Wt) {
  int idx = blockIdx.x * blockDim.x + threadIdx.x;
  if (idx >= D_ * D_) return;
  int k = idx / D_;
  int n = idx % D_;
  Wt[(size_t)n * D_ + k] = (__bf16)W[idx];
}

// ---------------------------------------------------------------------------
// Kernel 3: WMMA GEMM  C[4096,512] = A[4096,512](bf16) * Bt[512,512](bf16)
// Block = 128 thr (4 waves): 16M x 64N tile.
//   A tile (16x512)  -> LDS via TDM (wave 0, TENSORcnt, HW row-padding)
//   B tile (64x512)  -> LDS via global_load_async_to_lds_b128 (ASYNCcnt)
//   K loop: ds_load fragments + v_wmma_f32_16x16x32_bf16
// MODE 0: f32 +bias | 1: sigmoid f32 | 2: bf16 | 3: f32 +bias +residual
// ---------------------------------------------------------------------------
template <int MODE>
__global__ void gemm_kernel(const __bf16* __restrict__ A, const __bf16* __restrict__ Bt,
                            const float* __restrict__ bias,
                            float* __restrict__ outF, __bf16* __restrict__ outH,
                            const float* __restrict__ resid) {
  __shared__ __attribute__((aligned(16))) __bf16 ldsA[16 * KPAD];   // 16.25 KB
  __shared__ __attribute__((aligned(16))) __bf16 ldsB[64 * KPAD];   // 65 KB

  int lane  = threadIdx.x & 31;
  int wave  = threadIdx.x >> 5;
  int mtile = blockIdx.x * 16;
  int nt64  = blockIdx.y * 64;
  int hl    = (lane >= 16) ? 1 : 0;
  int l15   = lane & 15;

  // --- TDM DMA: A tile (16 rows x 512 bf16) with padded rows
  if (wave == 0) {
    tdm_load_tile16(lds_off_u32(ldsA), A + (size_t)mtile * D_);
  }

  // --- async copy: B tile (64 rows x 512 bf16) in 16B chunks, padded rows
  {
    const __bf16* bsrc = Bt + (size_t)nt64 * D_;
    unsigned bbase = lds_off_u32(ldsB);
    for (int c = threadIdx.x; c < 64 * 64; c += 128) {   // 4096 x 16B
      int row  = c >> 6;
      int col8 = (c & 63) * 8;
      async_copy_b128(bbase + (unsigned)(row * KPAD + col8) * 2,
                      bsrc + (size_t)row * D_ + col8);
    }
    async_wait0();
  }
  __syncthreads();

  const __bf16* la = ldsA + l15 * KPAD;
  const __bf16* lb = ldsB + (wave * 16 + l15) * KPAD;

  v8f acc = {};
#pragma unroll 4
  for (int k0 = 0; k0 < D_; k0 += 32) {
    union { unsigned u[8]; v16bf v; } af, bf;
#pragma unroll
    for (int i = 0; i < 8; ++i) {
      int ka = k0 + ((i < 4) ? 2 * i : 2 * i + 8) + hl * 8;   // A: 16-bit 16x32 layout
      af.u[i] = *(const unsigned*)(la + ka);
      int kb = k0 + ((i < 4) ? 2 * i : 2 * i + 8) + hl * 8;   // B: same k order as A
      bf.u[i] = *(const unsigned*)(lb + kb);
    }
    acc = __builtin_amdgcn_wmma_f32_16x16x32_bf16(false, af.v, false, bf.v,
                                                  (short)0, acc, false, false);
  }

  int ncol = nt64 + wave * 16 + l15;
  float bv = bias[ncol];
#pragma unroll
  for (int r = 0; r < 8; ++r) {
    int m = mtile + r + hl * 8;
    size_t o = (size_t)m * D_ + ncol;
    float val = acc[r] + bv;
    if (MODE == 0)      outF[o] = val;
    else if (MODE == 1) outF[o] = 1.0f / (1.0f + __expf(-val));
    else if (MODE == 2) outH[o] = (__bf16)val;
    else                outF[o] = val + resid[o];
  }
}

// ---------------------------------------------------------------------------
// Kernel 4: anchor softmax column stats over S (axis=1), wave per (b,d) column
// ---------------------------------------------------------------------------
__global__ void colstat_kernel(const float* __restrict__ fa,
                               float* __restrict__ cmax, float* __restrict__ csum) {
  int wave = (blockIdx.x * blockDim.x + threadIdx.x) >> 5;
  int lane = threadIdx.x & 31;
  if (wave >= B_ * D_) return;
  int b = wave / D_, d = wave % D_;
  const float* base = fa + (size_t)b * S_ * D_ + d;
  float m = -1e30f;
  for (int s = lane; s < S_; s += 32) m = fmaxf(m, base[(size_t)s * D_]);
#pragma unroll
  for (int off = 16; off > 0; off >>= 1) m = fmaxf(m, __shfl_xor(m, off, 32));
  float sum = 0.f;
  for (int s = lane; s < S_; s += 32) sum += __expf(base[(size_t)s * D_] - m);
#pragma unroll
  for (int off = 16; off > 0; off >>= 1) sum += __shfl_xor(sum, off, 32);
  if (lane == 0) { cmax[wave] = m; csum[wave] = sum; }
}

// ---------------------------------------------------------------------------
// Kernel 5: k_weighted (bf16) + V transpose to [b*D_ + d][s] (bf16)
// ---------------------------------------------------------------------------
__global__ void kw_vt_kernel(const float* __restrict__ kf, const float* __restrict__ fa,
                             const float* __restrict__ cmax, const float* __restrict__ csum,
                             const __bf16* __restrict__ vb,
                             __bf16* __restrict__ kw, __bf16* __restrict__ vt) {
  int idx = blockIdx.x * blockDim.x + threadIdx.x;
  if (idx >= NTOK * D_) return;
  int row = idx >> 9;          // token
  int d   = idx & 511;
  int b   = row >> 10;
  int s   = row & 1023;
  int col = b * D_ + d;
  float w = __expf(fa[idx] - cmax[col]) / csum[col];
  kw[idx] = (__bf16)(kf[idx] * w);
  vt[(size_t)col * S_ + s] = vb[idx];
}

// ---------------------------------------------------------------------------
// Kernel 6: streaming attention with head-axis softmax.
// Block = 512 threads = 16 waves; wave h owns head h; block owns (b, 16-query tile).
// Per 32-key step: 2 WMMAs (scores), exp, cross-head sum via ds_add_f32,
// normalize, repack probs via LDS into A-layout, 2 WMMAs (P@V accumulate).
// ---------------------------------------------------------------------------
__global__ void attn_kernel(const __bf16* __restrict__ qb, const __bf16* __restrict__ kw,
                            const __bf16* __restrict__ vt,
                            const float* __restrict__ mg, const float* __restrict__ hf,
                            __bf16* __restrict__ ob) {
  __shared__ __attribute__((aligned(16))) float  sums[2][16][16];     // 2 KB
  __shared__ __attribute__((aligned(16))) __bf16 probs[16][16][32];   // 16 KB

  int lane = threadIdx.x & 31;
  int h    = threadIdx.x >> 5;          // head = wave id
  int b    = blockIdx.x >> 6;           // 64 query-tiles per batch
  int s0   = (blockIdx.x & 63) * 16;
  int hl   = (lane >= 16) ? 1 : 0;
  int l15  = lane & 15;

  // Q fragment (A layout), kept in registers for the whole key loop
  union { unsigned u[8]; v16bf v; } qa;
  const __bf16* qrow = qb + (size_t)(b * S_ + s0 + l15) * D_ + h * HD_;
#pragma unroll
  for (int i = 0; i < 8; ++i) {
    int ka = ((i < 4) ? 2 * i : 2 * i + 8) + hl * 8;
    qa.u[i] = *(const unsigned*)(qrow + ka);
  }

  v8f acc0 = {}, acc1 = {};
  const __bf16* kwbase = kw + (size_t)(b * S_) * D_ + h * HD_;
  const __bf16* vt0 = vt + (size_t)(b * D_ + h * HD_ + l15) * S_;
  const __bf16* vt1 = vt0 + (size_t)16 * S_;

  for (int t0 = 0; t0 < S_; t0 += 32) {
    ((float*)sums)[threadIdx.x] = 0.f;    // 512 floats, 512 threads
    __syncthreads();

    // ---- scores for this head: q [16x32] x kw [32x16] (x2 key sub-blocks)
    union { unsigned u[8]; v16bf v; } bk0, bk1;
#pragma unroll
    for (int i = 0; i < 8; ++i) {
      int kk = ((i < 4) ? 2 * i : 2 * i + 8) + hl * 8;
      bk0.u[i] = *(const unsigned*)(kwbase + (size_t)(t0 + l15) * D_ + kk);
      bk1.u[i] = *(const unsigned*)(kwbase + (size_t)(t0 + 16 + l15) * D_ + kk);
    }
    v8f z = {};
    v8f sc0 = __builtin_amdgcn_wmma_f32_16x16x32_bf16(false, qa.v, false, bk0.v,
                                                      (short)0, z, false, false);
    v8f sc1 = __builtin_amdgcn_wmma_f32_16x16x32_bf16(false, qa.v, false, bk1.v,
                                                      (short)0, z, false, false);

    // ---- exp + cross-head reduction (softmax is over the HEAD axis)
    float e0[8], e1[8];
#pragma unroll
    for (int r = 0; r < 8; ++r) {
      e0[r] = __expf(sc0[r] * ATT_SCALE);
      e1[r] = __expf(sc1[r] * ATT_SCALE);
      int m = r + hl * 8;
      atomicAdd(&sums[0][m][l15], e0[r]);
      atomicAdd(&sums[1][m][l15], e1[r]);
    }
    __syncthreads();

    // ---- normalize; write probs as [m][t] bf16 for A-layout repack
#pragma unroll
    for (int r = 0; r < 8; ++r) {
      int m = r + hl * 8;
      probs[h][m][l15]      = (__bf16)(e0[r] / sums[0][m][l15]);
      probs[h][m][16 + l15] = (__bf16)(e1[r] / sums[1][m][l15]);
    }

    // ---- P @ V accumulate: probs [16x32] x v [32x16] (x2 d-halves)
    union { unsigned u[8]; v16bf v; } pa, bv0, bv1;
#pragma unroll
    for (int i = 0; i < 8; ++i) {
      int ka = ((i < 4) ? 2 * i : 2 * i + 8) + hl * 8;
      pa.u[i] = *(const unsigned*)(&probs[h][l15][ka]);   // wave-local RAW via dscnt
      int kt = t0 + ((i < 4) ? 2 * i : 2 * i + 8) + hl * 8;
      bv0.u[i] = *(const unsigned*)(vt0 + kt);
      bv1.u[i] = *(const unsigned*)(vt1 + kt);
    }
    acc0 = __builtin_amdgcn_wmma_f32_16x16x32_bf16(false, pa.v, false, bv0.v,
                                                   (short)0, acc0, false, false);
    acc1 = __builtin_amdgcn_wmma_f32_16x16x32_bf16(false, pa.v, false, bv1.v,
                                                   (short)0, acc1, false, false);
    __syncthreads();   // protect sums before next iteration re-zeroes
  }

  // ---- epilogue: out = attn * mg + hf  (bf16, feeds the Wo GEMM)
#pragma unroll
  for (int r = 0; r < 8; ++r) {
    int m = r + hl * 8;
    size_t row = (size_t)(b * S_ + s0 + m) * D_;
    int d0 = h * HD_ + l15;
    int d1 = d0 + 16;
    ob[row + d0] = (__bf16)(acc0[r] * mg[row + d0] + hf[row + d0]);
    ob[row + d1] = (__bf16)(acc1[r] * mg[row + d1] + hf[row + d1]);
  }
}

// ---------------------------------------------------------------------------
// Host-side launch
// ---------------------------------------------------------------------------
extern "C" void kernel_launch(void* const* d_in, const int* in_sizes, int n_in,
                              void* d_out, int out_size, void* d_ws, size_t ws_size,
                              hipStream_t stream) {
  const float* x    = (const float*)d_in[0];
  const float* ln_g = (const float*)d_in[1];
  const float* ln_b = (const float*)d_in[2];
  const float* W[7]  = {(const float*)d_in[3],  (const float*)d_in[5],
                        (const float*)d_in[7],  (const float*)d_in[9],
                        (const float*)d_in[11], (const float*)d_in[13],
                        (const float*)d_in[15]};
  const float* bias[7] = {(const float*)d_in[4],  (const float*)d_in[6],
                          (const float*)d_in[8],  (const float*)d_in[10],
                          (const float*)d_in[12], (const float*)d_in[14],
                          (const float*)d_in[16]};
  float* out = (float*)d_out;

  // ---- carve workspace
  char* ws = (char*)d_ws;
  size_t off = 0;
  auto carve = [&](size_t bytes) -> void* {
    void* p = ws + off;
    off += (bytes + 255) & ~(size_t)255;
    return p;
  };
  __bf16* xnb = (__bf16*)carve((size_t)NTOK * D_ * 2);
  __bf16* Wt[7];
  for (int i = 0; i < 7; ++i) Wt[i] = (__bf16*)carve((size_t)D_ * D_ * 2);
  float*  fa  = (float*) carve((size_t)NTOK * D_ * 4);
  float*  hfb = (float*) carve((size_t)NTOK * D_ * 4);
  float*  mgb = (float*) carve((size_t)NTOK * D_ * 4);
  float*  kf  = (float*) carve((size_t)NTOK * D_ * 4);
  __bf16* qb  = (__bf16*)carve((size_t)NTOK * D_ * 2);
  __bf16* vb  = (__bf16*)carve((size_t)NTOK * D_ * 2);
  __bf16* kw  = (__bf16*)carve((size_t)NTOK * D_ * 2);
  __bf16* vt  = (__bf16*)carve((size_t)NTOK * D_ * 2);
  __bf16* ob  = (__bf16*)carve((size_t)NTOK * D_ * 2);
  float*  cmax = (float*)carve((size_t)B_ * D_ * 4);
  float*  csum = (float*)carve((size_t)B_ * D_ * 4);
  (void)ws_size; (void)in_sizes; (void)n_in; (void)out_size;

  // 1) LayerNorm -> bf16
  ln_kernel<<<NTOK / 8, 256, 0, stream>>>(x, ln_g, ln_b, xnb);

  // 2) weights -> bf16 transposed
  for (int i = 0; i < 7; ++i)
    wconv_kernel<<<(D_ * D_) / 256, 256, 0, stream>>>(W[i], Wt[i]);

  dim3 ggrid(NTOK / 16, D_ / 64);
  // 3) six projections
  gemm_kernel<0><<<ggrid, 128, 0, stream>>>(xnb, Wt[0], bias[0], fa,  nullptr, nullptr); // fa
  gemm_kernel<0><<<ggrid, 128, 0, stream>>>(xnb, Wt[1], bias[1], hfb, nullptr, nullptr); // hf
  gemm_kernel<1><<<ggrid, 128, 0, stream>>>(xnb, Wt[2], bias[2], mgb, nullptr, nullptr); // mg
  gemm_kernel<2><<<ggrid, 128, 0, stream>>>(xnb, Wt[3], bias[3], nullptr, qb, nullptr);  // q
  gemm_kernel<0><<<ggrid, 128, 0, stream>>>(xnb, Wt[4], bias[4], kf,  nullptr, nullptr); // k
  gemm_kernel<2><<<ggrid, 128, 0, stream>>>(xnb, Wt[5], bias[5], nullptr, vb, nullptr);  // v

  // 4) anchor softmax stats over sequence axis
  colstat_kernel<<<(B_ * D_) / 8, 256, 0, stream>>>(fa, cmax, csum);

  // 5) k_weighted + v transpose
  kw_vt_kernel<<<(NTOK * D_) / 256, 256, 0, stream>>>(kf, fa, cmax, csum, vb, kw, vt);

  // 6) streaming attention (head-axis softmax), epilogue attn*mg + hf
  attn_kernel<<<B_ * (S_ / 16), 512, 0, stream>>>(qb, kw, vt, mgb, hfb, ob);

  // 7) output projection + residual -> d_out (f32)
  gemm_kernel<3><<<ggrid, 128, 0, stream>>>(ob, Wt[6], bias[6], out, nullptr, x);
}
